// FP8Experts_5257039970709
// MI455X (gfx1250) — 8-bit matrix operand order repaired and checked by reading, compiled, not run on hardware
//
#include <hip/hip_runtime.h>
#include <hip/hip_bf16.h>
#include <stdint.h>

#define HDIM   1024
#define IDIM   512
#define NEXP   16
#define GQ     128
#define CAP    2048
#define MTILE  32
#define FP8MAX 448.0f

typedef __attribute__((ext_vector_type(16))) int   v16i;
typedef __attribute__((ext_vector_type(8)))  float v8f;
typedef __attribute__((ext_vector_type(4)))  int   v4i;

#if defined(__has_builtin)
#if __has_builtin(__builtin_amdgcn_cvt_pk_fp8_f32)
#define USE_HW_FP8 1
#endif
#if __has_builtin(__builtin_amdgcn_global_load_async_to_lds_b128) && \
    __has_builtin(__builtin_amdgcn_s_wait_asynccnt)
#define USE_ASYNC_LDS 1
#endif
#endif

#ifdef USE_ASYNC_LDS
typedef __attribute__((address_space(1))) v4i gv4i_t;
typedef __attribute__((address_space(3))) v4i lv4i_t;
#endif

__device__ __forceinline__ void stage16(const unsigned char* gsrc, unsigned char* ldst) {
#ifdef USE_ASYNC_LDS
  __builtin_amdgcn_global_load_async_to_lds_b128((gv4i_t*)gsrc, (lv4i_t*)ldst, 0, 0);
#else
  *(uint4*)ldst = *(const uint4*)gsrc;
#endif
}
__device__ __forceinline__ void stage_wait() {
#ifdef USE_ASYNC_LDS
  __builtin_amdgcn_s_wait_asynccnt(0);
#endif
}

__device__ __forceinline__ unsigned char fp8_sw(float xf) {
  xf = fminf(FP8MAX, fmaxf(-FP8MAX, xf));
  unsigned u = __float_as_uint(xf);
  unsigned sign = (u >> 24) & 0x80u;
  u &= 0x7FFFFFFFu;
  if (u < 0x3C800000u) {
    int m = (int)rintf(__uint_as_float(u) * 512.0f);
    if (m > 7) return (unsigned char)(sign | 0x08u);
    return (unsigned char)(sign | (unsigned)m);
  }
  unsigned exp = u >> 23;
  unsigned man = u & 0x7FFFFFu;
  unsigned lsb = (man >> 20) & 1u;
  man += 0x7FFFFu + lsb;
  if (man >> 23) { man = 0; exp += 1; }
  int e8 = (int)exp - 127 + 7;
  unsigned m3 = (man >> 20) & 7u;
  if (e8 > 15 || (e8 == 15 && m3 > 6)) return (unsigned char)(sign | 0x7Eu);
  return (unsigned char)(sign | ((unsigned)e8 << 3) | m3);
}

__device__ __forceinline__ unsigned pack4_fp8(float a, float b, float c, float d) {
#ifdef USE_HW_FP8
  int v = __builtin_amdgcn_cvt_pk_fp8_f32(a, b, 0, false);
  v = __builtin_amdgcn_cvt_pk_fp8_f32(c, d, v, true);
  return (unsigned)v;
#else
  return (unsigned)fp8_sw(a) | ((unsigned)fp8_sw(b) << 8) |
         ((unsigned)fp8_sw(c) << 16) | ((unsigned)fp8_sw(d) << 24);
#endif
}

__device__ __forceinline__ v16i load_a_frag(const unsigned char* base, int hi) {
  union { v16i v; unsigned long long u[8]; } r;
#pragma unroll
  for (int p = 0; p < 4; ++p) {
    r.u[p]     = *(const unsigned long long*)(base + (p >> 1) * 32 + hi * 16 + (p & 1) * 8);
    r.u[p + 4] = *(const unsigned long long*)(base + 64 + (p >> 1) * 32 + hi * 16 + (p & 1) * 8);
  }
  return r.v;
}
__device__ __forceinline__ v16i load_b_frag(const unsigned char* base, int hi) {
  union { v16i v; v4i q[4]; } r;
#pragma unroll
  for (int g = 0; g < 4; ++g)
    r.q[g] = *(const v4i*)(base + g * 32 + hi * 16);
  return r.v;
}

__global__ void k_cvt_fp8(const float* __restrict__ src, unsigned* __restrict__ dst, int n4) {
  int i = blockIdx.x * blockDim.x + threadIdx.x;
  if (i >= n4) return;
  float4 f = ((const float4*)src)[i];
  dst[i] = pack4_fp8(fminf(FP8MAX, fmaxf(-FP8MAX, f.x)),
                     fminf(FP8MAX, fmaxf(-FP8MAX, f.y)),
                     fminf(FP8MAX, fmaxf(-FP8MAX, f.z)),
                     fminf(FP8MAX, fmaxf(-FP8MAX, f.w)));
}

__global__ void __launch_bounds__(128)
k_act_quant(const float* __restrict__ x, unsigned char* __restrict__ qx,
            float* __restrict__ sx) {
  const int t = blockIdx.x;
  const int tid = threadIdx.x;
  __shared__ float lmax[128];
  __shared__ float gscale[8];
  float4 a = ((const float4*)(x + (size_t)t * HDIM))[tid * 2 + 0];
  float4 b = ((const float4*)(x + (size_t)t * HDIM))[tid * 2 + 1];
  float m = fmaxf(fmaxf(fabsf(a.x), fabsf(a.y)), fmaxf(fabsf(a.z), fabsf(a.w)));
  m = fmaxf(m, fmaxf(fmaxf(fabsf(b.x), fabsf(b.y)), fmaxf(fabsf(b.z), fabsf(b.w))));
  lmax[tid] = m;
  __syncthreads();
  if (tid < 8) {
    float am = 0.f;
#pragma unroll
    for (int j = 0; j < 16; ++j) am = fmaxf(am, lmax[tid * 16 + j]);
    float sc = fmaxf(am, 1e-10f) / FP8MAX;
    gscale[tid] = sc;
    sx[(size_t)t * 8 + tid] = sc;
  }
  __syncthreads();
  float inv = 1.0f / gscale[tid >> 4];
  float f0 = fminf(FP8MAX, fmaxf(-FP8MAX, a.x * inv));
  float f1 = fminf(FP8MAX, fmaxf(-FP8MAX, a.y * inv));
  float f2 = fminf(FP8MAX, fmaxf(-FP8MAX, a.z * inv));
  float f3 = fminf(FP8MAX, fmaxf(-FP8MAX, a.w * inv));
  float f4 = fminf(FP8MAX, fmaxf(-FP8MAX, b.x * inv));
  float f5 = fminf(FP8MAX, fmaxf(-FP8MAX, b.y * inv));
  float f6 = fminf(FP8MAX, fmaxf(-FP8MAX, b.z * inv));
  float f7 = fminf(FP8MAX, fmaxf(-FP8MAX, b.w * inv));
  unsigned* dq = (unsigned*)qx;
  dq[(size_t)t * 256 + tid * 2 + 0] = pack4_fp8(f0, f1, f2, f3);
  dq[(size_t)t * 256 + tid * 2 + 1] = pack4_fp8(f4, f5, f6, f7);
}

__global__ void k_route(const int* __restrict__ idx, const float* __restrict__ w,
                        int* __restrict__ cnt, int* __restrict__ list,
                        float* __restrict__ wlist, int n, int tk) {
  int i = blockIdx.x * blockDim.x + threadIdx.x;
  if (i >= n) return;
  int e = idx[i];
  int t = i / tk;
  int pos = atomicAdd(&cnt[e], 1);
  if (pos < CAP) {
    list[e * CAP + pos] = t;
    wlist[e * CAP + pos] = w[i];
  }
}

__global__ void __launch_bounds__(256)
k_gateup(const unsigned char* __restrict__ qx, const float* __restrict__ sx,
         const unsigned char* __restrict__ qwgu, const float* __restrict__ sgu,
         const int* __restrict__ cnt, const int* __restrict__ list,
         unsigned char* __restrict__ qh, float* __restrict__ sh) {
  const int e = blockIdx.z;
  const int tile = blockIdx.y;
  const int colg = blockIdx.x;
  const int nrows = cnt[e];
  if (tile * MTILE >= nrows) return;

  __shared__ int           s_tok[MTILE];
  __shared__ float         s_sx[MTILE][8];
  __shared__ unsigned char s_a[MTILE * HDIM];
  __shared__ float         s_h[MTILE][132];
  __shared__ float         s_scale[MTILE];

  const int tid = threadIdx.x;
  if (tid < MTILE) {
    int slot = tile * MTILE + tid;
    s_tok[tid] = (slot < nrows) ? list[e * CAP + slot] : list[e * CAP];
  }
  __syncthreads();
  {
    int m = tid >> 3, kb = tid & 7;
    s_sx[m][kb] = sx[(size_t)s_tok[m] * 8 + kb];
  }
#pragma unroll
  for (int st = 0; st < (MTILE * HDIM) / (256 * 16); ++st) {
    int ofs = (st * 256 + tid) * 16;
    int row = ofs >> 10, col = ofs & (HDIM - 1);
    stage16(qx + (size_t)s_tok[row] * HDIM + col, s_a + ofs);
  }
  stage_wait();
  __syncthreads();

  const int wave = tid >> 5;
  const int lane = tid & 31;
  const int hi = lane >> 4;
  const int nn = lane & 15;
  const int og = colg * 128 + wave * 16 + nn;
  const int ou = og + 512;
  const int obg = colg;
  const int obu = colg + 4;

  float accg0[8], accg1[8], accu0[8], accu1[8];
#pragma unroll
  for (int r = 0; r < 8; ++r) { accg0[r] = accg1[r] = accu0[r] = accu1[r] = 0.f; }

  const unsigned char* a0 = s_a + (size_t)(lane & 15) * HDIM;
  const unsigned char* a1 = s_a + (size_t)((lane & 15) + 16) * HDIM;
  const unsigned char* brow_g = qwgu + ((size_t)(e * 1024 + og)) * HDIM;
  const unsigned char* brow_u = qwgu + ((size_t)(e * 1024 + ou)) * HDIM;
  const v8f zero = {0.f, 0.f, 0.f, 0.f, 0.f, 0.f, 0.f, 0.f};

  for (int kb = 0; kb < 8; ++kb) {
    const int ko = kb * 128;
    __builtin_prefetch(brow_g + ko + 128, 0, 0);
    v16i Bg = load_b_frag(brow_g + ko, hi);
    v16i Bu = load_b_frag(brow_u + ko, hi);
    v16i A0 = load_a_frag(a0 + ko, hi);
    v16i A1 = load_a_frag(a1 + ko, hi);
    v8f dg0 = __builtin_amdgcn_wmma_f32_16x16x128_fp8_fp8(A0, Bg, (short)0, zero, false, false);
    v8f du0 = __builtin_amdgcn_wmma_f32_16x16x128_fp8_fp8(A0, Bu, (short)0, zero, false, false);
    v8f dg1 = __builtin_amdgcn_wmma_f32_16x16x128_fp8_fp8(A1, Bg, (short)0, zero, false, false);
    v8f du1 = __builtin_amdgcn_wmma_f32_16x16x128_fp8_fp8(A1, Bu, (short)0, zero, false, false);
    float swg = sgu[((size_t)e * 8 + obg) * 8 + kb];
    float swu = sgu[((size_t)e * 8 + obu) * 8 + kb];
#pragma unroll
    for (int r = 0; r < 8; ++r) {
      int m = hi * 8 + r;
      float sa0 = s_sx[m][kb];
      float sa1 = s_sx[m + 16][kb];
      accg0[r] = fmaf(dg0[r], sa0 * swg, accg0[r]);
      accu0[r] = fmaf(du0[r], sa0 * swu, accu0[r]);
      accg1[r] = fmaf(dg1[r], sa1 * swg, accg1[r]);
      accu1[r] = fmaf(du1[r], sa1 * swu, accu1[r]);
    }
  }

#pragma unroll
  for (int r = 0; r < 8; ++r) {
    int m = hi * 8 + r;
    float g0 = accg0[r];
    float g1 = accg1[r];
    s_h[m][wave * 16 + nn]      = (g0 / (1.0f + __expf(-g0))) * accu0[r];
    s_h[m + 16][wave * 16 + nn] = (g1 / (1.0f + __expf(-g1))) * accu1[r];
  }
  __syncthreads();

  if (tid < MTILE) {
    float amax = 0.f;
#pragma unroll 8
    for (int c = 0; c < 128; ++c) amax = fmaxf(amax, fabsf(s_h[tid][c]));
    float sc = fmaxf(amax, 1e-10f) / FP8MAX;
    s_scale[tid] = sc;
    int slot = tile * MTILE + tid;
    sh[((size_t)e * CAP + slot) * 4 + colg] = sc;
  }
  __syncthreads();

  {
    int m = tid >> 3, seg = tid & 7;
    float inv = 1.0f / s_scale[m];
    float f[16];
#pragma unroll
    for (int j = 0; j < 16; ++j)
      f[j] = fminf(FP8MAX, fmaxf(-FP8MAX, s_h[m][seg * 16 + j] * inv));
    int slot = tile * MTILE + m;
    uint4 q;
    q.x = pack4_fp8(f[0], f[1], f[2], f[3]);
    q.y = pack4_fp8(f[4], f[5], f[6], f[7]);
    q.z = pack4_fp8(f[8], f[9], f[10], f[11]);
    q.w = pack4_fp8(f[12], f[13], f[14], f[15]);
    *(uint4*)(qh + ((size_t)e * CAP + slot) * IDIM + colg * 128 + seg * 16) = q;
  }
}

__global__ void __launch_bounds__(256)
k_down(const unsigned char* __restrict__ qh, const float* __restrict__ sh,
       const unsigned char* __restrict__ qwdn, const float* __restrict__ sdn,
       const int* __restrict__ cnt, const int* __restrict__ list,
       const float* __restrict__ wlist, float* __restrict__ out) {
  const int e = blockIdx.z;
  const int tile = blockIdx.y;
  const int hg = blockIdx.x;
  const int nrows = cnt[e];
  if (tile * MTILE >= nrows) return;

  __shared__ float         s_sh[MTILE][4];
  __shared__ unsigned char s_a[MTILE * IDIM];
  const int tid = threadIdx.x;
  if (tid < 128) {
    int m = tid >> 2, kb = tid & 3;
    int slot = tile * MTILE + m;
    s_sh[m][kb] = sh[((size_t)e * CAP + slot) * 4 + kb];
  }
  {
    const unsigned char* src = qh + ((size_t)e * CAP + tile * MTILE) * IDIM;
#pragma unroll
    for (int st = 0; st < (MTILE * IDIM) / (256 * 16); ++st) {
      int ofs = (st * 256 + tid) * 16;
      stage16(src + ofs, s_a + ofs);
    }
  }
  stage_wait();
  __syncthreads();

  const int wave = tid >> 5;
  const int lane = tid & 31;
  const int hi = lane >> 4;
  const int nn = lane & 15;
  const int ocol = hg * 128 + wave * 16 + nn;

  float acc0[8], acc1[8];
#pragma unroll
  for (int r = 0; r < 8; ++r) { acc0[r] = 0.f; acc1[r] = 0.f; }

  const unsigned char* a0 = s_a + (size_t)(lane & 15) * IDIM;
  const unsigned char* a1 = s_a + (size_t)((lane & 15) + 16) * IDIM;
  const unsigned char* brow = qwdn + ((size_t)(e * 1024 + ocol)) * IDIM;
  const v8f zero = {0.f, 0.f, 0.f, 0.f, 0.f, 0.f, 0.f, 0.f};

  for (int kb = 0; kb < 4; ++kb) {
    const int ko = kb * 128;
    v16i B  = load_b_frag(brow + ko, hi);
    v16i A0 = load_a_frag(a0 + ko, hi);
    v16i A1 = load_a_frag(a1 + ko, hi);
    v8f d0 = __builtin_amdgcn_wmma_f32_16x16x128_fp8_fp8(A0, B, (short)0, zero, false, false);
    v8f d1 = __builtin_amdgcn_wmma_f32_16x16x128_fp8_fp8(A1, B, (short)0, zero, false, false);
    float sw = sdn[((size_t)e * 8 + hg) * 4 + kb];
#pragma unroll
    for (int r = 0; r < 8; ++r) {
      int m = hi * 8 + r;
      acc0[r] = fmaf(d0[r], s_sh[m][kb] * sw, acc0[r]);
      acc1[r] = fmaf(d1[r], s_sh[m + 16][kb] * sw, acc1[r]);
    }
  }

#pragma unroll
  for (int r = 0; r < 8; ++r) {
    int m = hi * 8 + r;
    int slot0 = tile * MTILE + m;
    int slot1 = slot0 + 16;
    if (slot0 < nrows) {
      unsafeAtomicAdd(&out[(size_t)list[e * CAP + slot0] * HDIM + ocol],
                      acc0[r] * wlist[e * CAP + slot0]);
    }
    if (slot1 < nrows) {
      unsafeAtomicAdd(&out[(size_t)list[e * CAP + slot1] * HDIM + ocol],
                      acc1[r] * wlist[e * CAP + slot1]);
    }
  }
}

extern "C" void kernel_launch(void* const* d_in, const int* in_sizes, int n_in,
                              void* d_out, int out_size, void* d_ws, size_t ws_size,
                              hipStream_t stream) {
  const float* x        = (const float*)d_in[0];
  const int*   topk_idx = (const int*)d_in[1];
  const float* topk_w   = (const float*)d_in[2];
  const float* wgu      = (const float*)d_in[3];
  const float* sgu      = (const float*)d_in[4];
  const float* wdn      = (const float*)d_in[5];
  const float* sdn      = (const float*)d_in[6];
  float* out = (float*)d_out;

  const int T  = in_sizes[0] / HDIM;
  const int TK = in_sizes[1] / T;

  char* p = (char*)d_ws;
  auto carve = [&](size_t n) { char* r = p; p += (n + 255) & ~(size_t)255; return r; };
  unsigned char* qx   = (unsigned char*)carve((size_t)T * HDIM);
  float*         sx   = (float*)carve((size_t)T * 8 * sizeof(float));
  unsigned char* qwgu = (unsigned char*)carve((size_t)NEXP * 1024 * HDIM);
  unsigned char* qwdn = (unsigned char*)carve((size_t)NEXP * HDIM * IDIM);
  int*           cnt  = (int*)carve(NEXP * sizeof(int));
  int*           list = (int*)carve((size_t)NEXP * CAP * sizeof(int));
  float*         wls  = (float*)carve((size_t)NEXP * CAP * sizeof(float));
  unsigned char* qh   = (unsigned char*)carve((size_t)NEXP * CAP * IDIM);
  float*         sh   = (float*)carve((size_t)NEXP * CAP * 4 * sizeof(float));

  (void)hipMemsetAsync(cnt, 0, NEXP * sizeof(int), stream);
  (void)hipMemsetAsync(out, 0, (size_t)out_size * sizeof(float), stream);

  {
    int n4 = NEXP * 1024 * HDIM / 4;
    k_cvt_fp8<<<(n4 + 255) / 256, 256, 0, stream>>>(wgu, (unsigned*)qwgu, n4);
  }
  {
    int n4 = NEXP * HDIM * IDIM / 4;
    k_cvt_fp8<<<(n4 + 255) / 256, 256, 0, stream>>>(wdn, (unsigned*)qwdn, n4);
  }
  k_act_quant<<<T, 128, 0, stream>>>(x, qx, sx);
  {
    int n = T * TK;
    k_route<<<(n + 255) / 256, 256, 0, stream>>>(topk_idx, topk_w, cnt, list, wls, n, TK);
  }
  k_gateup<<<dim3(IDIM / 128, CAP / MTILE, NEXP), 256, 0, stream>>>(qx, sx, qwgu, sgu, cnt, list, qh, sh);
  k_down<<<dim3(HDIM / 128, CAP / MTILE, NEXP), 256, 0, stream>>>(qh, sh, qwdn, sdn, cnt, list, wls, out);
}
